// MCAT_75797582839970
// MI455X (gfx1250) — matrix operand order repaired and checked by reading, not compiled, not run on hardware
//
#include <hip/hip_runtime.h>
#include <hip/hip_bf16.h>
#include <math.h>

// ---------------------------------------------------------------------------
// MCAT fused pipeline for MI455X (gfx1250, wave32, WMMA bf16).
// fused = relu(X@W1+b1)@Wv+bv  (softmax over singleton axis == 1, q/k dead)
// gate  = tanh(fused@Wa+ba) * sigmoid(fused@Wb+bb)
// A_n   = gate_n . ac_w + ac_b ; pooled = softmax_N(A) @ fused ; MLP head.
// Weights are stored K-transposed in bf16 so B-fragments are contiguous-K
// (two ds_load_b128 per fragment); weight slabs staged to LDS with async
// global->LDS copies (ASYNCcnt) when the builtin is available.
// ---------------------------------------------------------------------------

typedef __attribute__((ext_vector_type(16))) __bf16 v16bf;
typedef __attribute__((ext_vector_type(8)))  __bf16 v8bf;
typedef __attribute__((ext_vector_type(4)))  __bf16 v4bf;
typedef __attribute__((ext_vector_type(8)))  float  v8f;
typedef __attribute__((ext_vector_type(4)))  int    v4i;

#define NTHREADS 256   // 8 waves (wave32)
#define MT  32         // rows per block
#define DH  256        // hidden dim
#define KT  32         // WMMA K step (16x16x32 bf16)
#define DIN 1024
#define DCLS 128
#define NCLS 4

#if defined(__has_builtin)
#  if __has_builtin(__builtin_amdgcn_global_load_async_to_lds_b128)
#    define USE_ASYNC_LDS 1
#  endif
#endif
#ifndef USE_ASYNC_LDS
#  define USE_ASYNC_LDS 0
#endif

#if USE_ASYNC_LDS
// global source: 128-bit payload pointer in the global (__device__) AS
#define GPTR(p) ((__attribute__((address_space(1))) v4i*)(uintptr_t)(p))
// LDS destination: 128-bit payload pointer in AS(3) (low 32 bits of generic)
#define LPTR(p) ((__attribute__((address_space(3))) v4i*)(uint32_t)(uintptr_t)(p))
__device__ __forceinline__ void wait_async0() {
#if __has_builtin(__builtin_amdgcn_s_wait_asynccnt)
    __builtin_amdgcn_s_wait_asynccnt(0);
#else
    asm volatile("s_wait_asynccnt 0x0" ::: "memory");
#endif
}
#endif

union Frag16 { v16bf v; v8bf h[2]; };

// --- f32 -> bf16 transposed weight conversion: dst[n][k] = src[k][n] -------
__global__ __launch_bounds__(256) void cvt_bf16_t_k(const float* __restrict__ src,
                                                    __bf16* __restrict__ dst,
                                                    int K, int Ncols) {
    int i = blockIdx.x * 256 + threadIdx.x;
    if (i < K * Ncols) {
        int k = i / Ncols, n = i % Ncols;
        dst[(size_t)n * K + k] = (__bf16)src[i];
    }
}

// Stage one K-slab of a transposed weight matrix into LDS: ldsWt[n][0..31].
__device__ __forceinline__ void stage_wt(__bf16 (&ldsWt)[DH][KT],
                                         const __bf16* __restrict__ Wt,
                                         int kk, int kstride, int tid) {
    for (int i = tid; i < DH * 4; i += NTHREADS) {   // 1024 chunks of 8 bf16
        int n = i >> 2, rc = (i & 3) * 8;
#if USE_ASYNC_LDS
        __builtin_amdgcn_global_load_async_to_lds_b128(
            GPTR(&Wt[(size_t)n * kstride + kk + rc]), LPTR(&ldsWt[n][rc]), 0, 0);
#else
        *(v8bf*)&ldsWt[n][rc] = *(const v8bf*)&Wt[(size_t)n * kstride + kk + rc];
#endif
    }
#if USE_ASYNC_LDS
    wait_async0();
#endif
}

// 32x256 += 32xK_slab * slab x 256 with A (bf16) in LDS, W transposed in LDS.
__device__ __forceinline__ void gemm_hid(v8f acc[4],
                                         const __bf16* __restrict__ Wt,  // DH x DH (n-major)
                                         __bf16 (&ldsA)[MT][DH],
                                         __bf16 (&ldsWt)[DH][KT],
                                         int tid, int mSub, int nBase,
                                         int l15, int half) {
    v8f z = {};
#pragma unroll
    for (int t = 0; t < 4; ++t) acc[t] = z;
    for (int kk = 0; kk < DH; kk += KT) {
        stage_wt(ldsWt, Wt, kk, DH, tid);
        __syncthreads();
        Frag16 af;   // A layout: lanes0-15 K{0..7,16..23}, lanes16-31 K{8..15,24..31}
        int arow = mSub * 16 + l15;
        af.h[0] = *(const v8bf*)&ldsA[arow][kk + half * 8];
        af.h[1] = *(const v8bf*)&ldsA[arow][kk + 16 + half * 8];
#pragma unroll
        for (int t = 0; t < 4; ++t) {
            Frag16 bf_;  // B layout: lanes0-15 K{0..7,16..23}, lanes16-31 K{8..15,24..31}
            int col = (nBase + t) * 16 + l15;
            bf_.h[0] = *(const v8bf*)&ldsWt[col][half * 8];
            bf_.h[1] = *(const v8bf*)&ldsWt[col][16 + half * 8];
            acc[t] = __builtin_amdgcn_wmma_f32_16x16x32_bf16(
                false, af.v, false, bf_.v, (short)0, acc[t], false, false);
        }
        __syncthreads();
    }
}

__global__ __launch_bounds__(NTHREADS) void mcat_main(
    const float* __restrict__ xp,                                     // N x 1024
    const __bf16* __restrict__ w1t, const float* __restrict__ b1,     // 256x1024 (n-major)
    const __bf16* __restrict__ wvt, const float* __restrict__ bv,     // 256x256
    const __bf16* __restrict__ wat, const float* __restrict__ ba,
    const __bf16* __restrict__ wbt, const float* __restrict__ bb,
    const float* __restrict__ acw, const float* __restrict__ acb,
    float* __restrict__ fusedOut,                                     // N x 256
    float* __restrict__ Ascore,                                       // N
    int N) {
    __shared__ __bf16 ldsH[MT][DH];     // h, later fused (bf16)   16 KB
    __shared__ __bf16 ldsG[MT][DH];     // tanh branch             16 KB
    __shared__ __bf16 ldsWt[DH][KT];    // weight K-slab (n-major) 16 KB
    __shared__ __bf16 ldsX[MT][KT];     // X K-slab                 2 KB
    __shared__ float  rowAcc[MT];

    const int tid  = threadIdx.x;
    const int wid  = tid >> 5;
    const int lane = tid & 31;
    const int l15  = lane & 15;
    const int half = lane >> 4;
    const int mSub = wid >> 2;          // 0..1
    const int nBase = (wid & 3) * 4;    // 4 n-subtiles of 16 cols
    const int n0 = blockIdx.x * MT;

    if (tid < MT) rowAcc[tid] = 0.f;

    // ---- GEMM1: h = relu(X @ W1 + b1), K = 1024 -------------------------
    v8f acc[4];
    { v8f z = {}; for (int t = 0; t < 4; ++t) acc[t] = z; }
    for (int kk = 0; kk < DIN; kk += KT) {
        if (kk + KT < DIN) {   // prefetch next X K-slab (one line per row)
            int pr = n0 + (tid & 31);
            if (pr < N) __builtin_prefetch(&xp[(size_t)pr * DIN + kk + KT], 0, 0);
        }
        // X tile: 32 rows x 32 cols f32 -> bf16; one 4-wide chunk per thread
        {
            int r = tid >> 3, c4 = (tid & 7) * 4;
            int n = n0 + r;
            float4 xv = make_float4(0.f, 0.f, 0.f, 0.f);
            if (n < N) xv = *(const float4*)&xp[(size_t)n * DIN + kk + c4];
            v4bf xb;
            xb[0] = (__bf16)xv.x; xb[1] = (__bf16)xv.y;
            xb[2] = (__bf16)xv.z; xb[3] = (__bf16)xv.w;
            *(v4bf*)&ldsX[r][c4] = xb;
        }
        stage_wt(ldsWt, w1t, kk, DIN, tid);
        __syncthreads();
        Frag16 af;
        int arow = mSub * 16 + l15;
        af.h[0] = *(const v8bf*)&ldsX[arow][half * 8];
        af.h[1] = *(const v8bf*)&ldsX[arow][16 + half * 8];
#pragma unroll
        for (int t = 0; t < 4; ++t) {
            Frag16 bf_;
            int col = (nBase + t) * 16 + l15;
            bf_.h[0] = *(const v8bf*)&ldsWt[col][half * 8];
            bf_.h[1] = *(const v8bf*)&ldsWt[col][16 + half * 8];
            acc[t] = __builtin_amdgcn_wmma_f32_16x16x32_bf16(
                false, af.v, false, bf_.v, (short)0, acc[t], false, false);
        }
        __syncthreads();
    }
    // epilogue: bias + relu -> ldsH (bf16). C layout: row = r + half*8, col = l15.
#pragma unroll
    for (int t = 0; t < 4; ++t) {
        int col = (nBase + t) * 16 + l15;
        float bias = b1[col];
#pragma unroll
        for (int r = 0; r < 8; ++r) {
            int row = mSub * 16 + half * 8 + r;
            float hv = acc[t][r] + bias;
            ldsH[row][col] = (__bf16)fmaxf(hv, 0.f);
        }
    }
    __syncthreads();

    // ---- GEMM2: fused = h @ Wv + bv -------------------------------------
    gemm_hid(acc, wvt, ldsH, ldsWt, tid, mSub, nBase, l15, half);
#pragma unroll
    for (int t = 0; t < 4; ++t) {
        int col = (nBase + t) * 16 + l15;
        float bias = bv[col];
#pragma unroll
        for (int r = 0; r < 8; ++r) {
            int row = mSub * 16 + half * 8 + r;
            float f = acc[t][r] + bias;
            int n = n0 + row;
            if (n < N) fusedOut[(size_t)n * DH + col] = f;
            ldsH[row][col] = (__bf16)f;   // fused replaces h (all reads done)
        }
    }
    __syncthreads();

    // ---- GEMM3: tanh branch ---------------------------------------------
    gemm_hid(acc, wat, ldsH, ldsWt, tid, mSub, nBase, l15, half);
#pragma unroll
    for (int t = 0; t < 4; ++t) {
        int col = (nBase + t) * 16 + l15;
        float bias = ba[col];
#pragma unroll
        for (int r = 0; r < 8; ++r) {
            int row = mSub * 16 + half * 8 + r;
            ldsG[row][col] = (__bf16)tanhf(acc[t][r] + bias);
        }
    }
    __syncthreads();

    // ---- GEMM4: sigmoid branch, gate, attention-score GEMV --------------
    gemm_hid(acc, wbt, ldsH, ldsWt, tid, mSub, nBase, l15, half);
#pragma unroll
    for (int t = 0; t < 4; ++t) {
        int col = (nBase + t) * 16 + l15;
        float bias = bb[col];
        float wc = acw[col];
#pragma unroll
        for (int r = 0; r < 8; ++r) {
            int row = mSub * 16 + half * 8 + r;
            float sg = 1.f / (1.f + __expf(-(acc[t][r] + bias)));
            float gate = sg * (float)ldsG[row][col];
            atomicAdd(&rowAcc[row], gate * wc);   // ds_add_f32
        }
    }
    __syncthreads();
    if (tid < MT) {
        int n = n0 + tid;
        if (n < N) Ascore[n] = rowAcc[tid] + acb[0];
    }
}

// ---- global max of attention scores (single block, deterministic) --------
__global__ __launch_bounds__(256) void reduce_max_k(const float* __restrict__ A,
                                                    int N, float* __restrict__ stats) {
    __shared__ float sm[256];
    float m = -3.0e38f;
    for (int i = threadIdx.x; i < N; i += 256) m = fmaxf(m, A[i]);
    sm[threadIdx.x] = m;
    __syncthreads();
    for (int s = 128; s > 0; s >>= 1) {
        if (threadIdx.x < s) sm[threadIdx.x] = fmaxf(sm[threadIdx.x], sm[threadIdx.x + s]);
        __syncthreads();
    }
    if (threadIdx.x == 0) stats[0] = sm[0];
}

// ---- per-block partial softmax pooling (deterministic, no float atomics) -
__global__ __launch_bounds__(256) void pool_partial_k(const float* __restrict__ A,
                                                      const float* __restrict__ fused,
                                                      const float* __restrict__ stats,
                                                      float* __restrict__ partials,  // nblk x 257
                                                      int N) {
    __shared__ float wloc[256];
    const int t = threadIdx.x;
    const int n0 = blockIdx.x * 256;
    const int n = n0 + t;
    float mx = stats[0];
    float w = (n < N) ? __expf(A[n] - mx) : 0.f;
    wloc[t] = w;
    __syncthreads();
    int rows = N - n0; if (rows > 256) rows = 256;
    float acc = 0.f;                      // this thread owns output dim t
    for (int r = 0; r < rows; ++r)
        acc += wloc[r] * fused[(size_t)(n0 + r) * DH + t];
    partials[(size_t)blockIdx.x * 257 + t] = acc;
    __syncthreads();
    for (int s = 128; s > 0; s >>= 1) {
        if (t < s) wloc[t] += wloc[t + s];
        __syncthreads();
    }
    if (t == 0) partials[(size_t)blockIdx.x * 257 + 256] = wloc[0];
}

// ---- finalize: reduce partials, normalize, classifier MLP ----------------
__global__ __launch_bounds__(256) void finalize_k(const float* __restrict__ partials,
                                                  int nblk,
                                                  const float* __restrict__ c1w,
                                                  const float* __restrict__ c1b,
                                                  const float* __restrict__ c2w,
                                                  const float* __restrict__ c2b,
                                                  float* __restrict__ out) {
    __shared__ float p[DH + 1];
    __shared__ float h[DCLS];
    const int t = threadIdx.x;
    float s = 0.f;
    for (int b = 0; b < nblk; ++b) s += partials[(size_t)b * 257 + t];
    p[t] = s;
    if (t == 0) {
        float d = 0.f;
        for (int b = 0; b < nblk; ++b) d += partials[(size_t)b * 257 + 256];
        p[DH] = d;
    }
    __syncthreads();
    float inv = 1.f / p[DH];
    if (t < DCLS) {
        float acc = c1b[t];
        for (int d = 0; d < DH; ++d) acc += p[d] * inv * c1w[(size_t)d * DCLS + t];
        h[t] = fmaxf(acc, 0.f);
    }
    __syncthreads();
    if (t < NCLS) {
        float o = c2b[t];
        for (int j = 0; j < DCLS; ++j) o += h[j] * c2w[(size_t)j * NCLS + t];
        out[t] = o;
    }
}

extern "C" void kernel_launch(void* const* d_in, const int* in_sizes, int n_in,
                              void* d_out, int out_size, void* d_ws, size_t ws_size,
                              hipStream_t stream) {
    const float* xp  = (const float*)d_in[0];
    const float* w1  = (const float*)d_in[2];
    const float* b1  = (const float*)d_in[3];
    const float* wv  = (const float*)d_in[10];
    const float* bv  = (const float*)d_in[11];
    const float* wa  = (const float*)d_in[12];
    const float* ba  = (const float*)d_in[13];
    const float* wb  = (const float*)d_in[14];
    const float* bb  = (const float*)d_in[15];
    const float* acw = (const float*)d_in[16];
    const float* acb = (const float*)d_in[17];
    const float* c1w = (const float*)d_in[18];
    const float* c1b = (const float*)d_in[19];
    const float* c2w = (const float*)d_in[20];
    const float* c2b = (const float*)d_in[21];
    float* out = (float*)d_out;

    const int N = in_sizes[0] / DIN;

    // workspace carve-out (~52.3 MB)
    char* ws = (char*)d_ws;
    size_t off = 0;
    auto carve = [&](size_t bytes) -> char* {
        char* p = ws + off;
        off = (off + bytes + 255) & ~(size_t)255;
        return p;
    };
    __bf16* w1t = (__bf16*)carve((size_t)DIN * DH * sizeof(__bf16));
    __bf16* wvt = (__bf16*)carve((size_t)DH * DH * sizeof(__bf16));
    __bf16* wat = (__bf16*)carve((size_t)DH * DH * sizeof(__bf16));
    __bf16* wbt = (__bf16*)carve((size_t)DH * DH * sizeof(__bf16));
    float* stats  = (float*)carve(16 * sizeof(float));
    float* Ascore = (float*)carve((size_t)N * sizeof(float));
    const int nPool = (N + 255) / 256;
    float* partials = (float*)carve((size_t)nPool * 257 * sizeof(float));
    float* fused = (float*)carve((size_t)N * DH * sizeof(float));

    cvt_bf16_t_k<<<(DIN * DH + 255) / 256, 256, 0, stream>>>(w1, w1t, DIN, DH);
    cvt_bf16_t_k<<<(DH * DH + 255) / 256, 256, 0, stream>>>(wv, wvt, DH, DH);
    cvt_bf16_t_k<<<(DH * DH + 255) / 256, 256, 0, stream>>>(wa, wat, DH, DH);
    cvt_bf16_t_k<<<(DH * DH + 255) / 256, 256, 0, stream>>>(wb, wbt, DH, DH);

    const int nblk = (N + MT - 1) / MT;
    mcat_main<<<nblk, NTHREADS, 0, stream>>>(xp, w1t, b1, wvt, bv, wat, ba,
                                             wbt, bb, acw, acb, fused, Ascore, N);
    reduce_max_k<<<1, 256, 0, stream>>>(Ascore, N, stats);
    pool_partial_k<<<nPool, 256, 0, stream>>>(Ascore, fused, stats, partials, N);
    finalize_k<<<1, 256, 0, stream>>>(partials, nPool, c1w, c1b, c2w, c2b, out);
}
